// Router_67310727462998
// MI455X (gfx1250) — matrix operand order repaired and checked by reading, not compiled, not run on hardware
//
#include <hip/hip_runtime.h>
#include <hip/hip_bf16.h>
#include <math.h>

// ---------------------------------------------------------------------------
// MI455X (gfx1250) fused MoE router: budget + GEMM1 + GEMM2 + softmax stats +
// top-32 + masked weights + aux-loss accumulators, scores kept in LDS only.
// Matrix math on v_wmma_f32_16x16x32_bf16 (wave32 WMMA). Fragment loads are
// depth-3 software pipelined; __builtin_amdgcn_sched_barrier(0) fences pin
// the prefetch loads ahead of each WMMA so SIInsertWaitcnts emits partial
// s_wait counts (loads stay in flight across ~3 WMMAs) instead of draining
// every counter to zero.
// ---------------------------------------------------------------------------

typedef __attribute__((ext_vector_type(16))) __bf16 v16bf;
typedef __attribute__((ext_vector_type(8)))  float  v8f;

union BFrag {
    v16bf v;
    uint4 q[2];
};

constexpr int kD = 1024;   // model dim
constexpr int kR = 128;    // router hidden
constexpr int kN = 4096;   // memory slots
constexpr int kM = 16;     // tokens per block (WMMA M tile)
constexpr int kMaxK = 32;  // top-k
constexpr int kTokens = 32768;
constexpr int kPipe = 3;   // software pipeline depth (4-slot ring)

// LDS layout (bytes). scores region aliases the x-bf16 staging region.
constexpr int kOffScores = 0;                       // 16*4096*4 = 262144
constexpr int kOffHid    = 262144;                  // 16*128*2  = 4096
constexpr int kOffRed    = 266240;                  // 256*4     = 1024
constexpr int kOffTmax   = 267264;                  // 16*4
constexpr int kOffTsum   = 267328;                  // 16*4
constexpr int kOffBud    = 267392;                  // 16*4
constexpr int kOffTopV   = 267456;                  // 16*32*4   = 2048
constexpr int kOffTopI   = 269504;                  // 16*32*4   = 2048
constexpr int kSmemBytes = 271552;                  // ~265 KB < 320 KB WGP LDS

__device__ __forceinline__ unsigned short f32_to_bf16_rne(float f) {
    unsigned int u = __float_as_uint(f);
    u += 0x7FFFu + ((u >> 16) & 1u);     // round-to-nearest-even
    return (unsigned short)(u >> 16);
}

// ---------------------------------------------------------------------------
// Prep: zero the global accumulators (avg_probs partials + counts).
// ---------------------------------------------------------------------------
__global__ void zero_accum(float* __restrict__ p, int n) {
    int i = blockIdx.x * 256 + threadIdx.x;
    if (i < n) p[i] = 0.0f;
}

// ---------------------------------------------------------------------------
// Prep: convert a row-major f32 weight matrix [K x ncols] (K = kcCount*32)
// into bf16 packed in this kernel's B-fragment order (contiguous K):
//   fragment f = colTile*kcCount + kc, element = lane*16 + h
//   lane<16 : N = colTile*16 + lane,      K = kc*32 + h        (h = 0..15)
//   lane>=16: N = colTile*16 + (lane-16), K = kc*32 + 16 + h
// Steady-state loads become two coalesced 16B loads per lane from L2.
// ---------------------------------------------------------------------------
__global__ void pack_b_frags(const float* __restrict__ W,
                             unsigned short* __restrict__ dst,
                             int ncols, int kcCount, int total) {
    int e = blockIdx.x * 256 + threadIdx.x;
    if (e >= total) return;
    int h    = e & 15;
    int lane = (e >> 4) & 31;
    int f    = e >> 9;
    int kc   = f % kcCount;
    int ct   = f / kcCount;
    int col  = ct * 16 + (lane & 15);
    int k    = kc * 32 + ((lane < 16) ? h : (16 + h));
    dst[e] = f32_to_bf16_rne(W[(size_t)k * ncols + col]);
}

// ---------------------------------------------------------------------------
// Main fused router kernel. One block = 16 tokens, 256 threads = 8 waves.
// ---------------------------------------------------------------------------
__global__ void __launch_bounds__(256)
router_main(const float* __restrict__ x,
            const float* __restrict__ W_c, const float* __restrict__ b_c,
            const float* __restrict__ b_h, const float* __restrict__ b_s,
            const unsigned short* __restrict__ WhP,   // packed bf16 W_h frags
            const unsigned short* __restrict__ WsP,   // packed bf16 W_s frags
            float* __restrict__ avg_accum,            // [N] prob sums
            float* __restrict__ counts,               // [N] selection counts
            int*   __restrict__ out_topi,             // [tokens*32]
            float* __restrict__ out_w,                // [tokens*32]
            int*   __restrict__ out_bud)              // [tokens]
{
    extern __shared__ char smem[];
    float*          scores = (float*)(smem + kOffScores);        // 16 x 4096 f32
    unsigned short* xbf    = (unsigned short*)(smem);            // aliases scores
    unsigned short* hid    = (unsigned short*)(smem + kOffHid);  // 16 x 128 bf16
    float*          red    = (float*)(smem + kOffRed);
    float*          tmax   = (float*)(smem + kOffTmax);
    float*          tsum   = (float*)(smem + kOffTsum);
    int*            budl   = (int*)  (smem + kOffBud);
    float*          topv   = (float*)(smem + kOffTopV);
    int*            topil  = (int*)  (smem + kOffTopI);

    const int tid  = threadIdx.x;
    const int lane = tid & 31;
    const int w    = tid >> 5;            // wave id 0..7
    const int t0   = blockIdx.x * kM;     // first token of this tile
    const float* xg = x + (size_t)t0 * kD;

    // ---- Phase 0: stream x tile (contiguous 64 KB) -> bf16 in LDS -----------
    for (int i = tid; i < kM * kD / 4; i += 256) {
        float4 v = ((const float4*)xg)[i];
        ushort4 h;
        h.x = f32_to_bf16_rne(v.x);
        h.y = f32_to_bf16_rne(v.y);
        h.z = f32_to_bf16_rne(v.z);
        h.w = f32_to_bf16_rne(v.w);
        ((ushort4*)xbf)[i] = h;
    }
    // Prefetch next block's x tile into cache (global_prefetch_b8).
    if (blockIdx.x + 1 < gridDim.x)
        __builtin_prefetch(xg + (size_t)kM * kD + (size_t)tid * 64, 0, 1);

    // ---- Phase 0b: complexity -> dynamic budget (x read hits L2) ------------
    {
        int tok = tid >> 4, sub = tid & 15;
        float p = 0.0f;
        for (int j = 0; j < kD / 16; ++j) {
            int idx = sub + j * 16;
            p += xg[(size_t)tok * kD + idx] * W_c[idx];
        }
        red[tid] = p;
    }
    __syncthreads();
    if (tid < kM) {
        float s = 0.0f;
        for (int i = 0; i < 16; ++i) s += red[tid * 16 + i];
        float z = s + b_c[0];
        float c = 1.0f / (1.0f + __expf(-z));
        int bud = (int)floorf(4.0f + 28.0f * c * c);   // MIN_K + (MAX-MIN)*c^2
        if (bud > 31) bud = 31;
        budl[tid] = bud;
        out_bud[t0 + tid] = bud;
    }
    __syncthreads();

    const int m     = lane & 15;
    const int aoff  = (lane < 16) ? 0 : 16;  // A-frag K sub-offset (contiguous K order)
    const int rbase = (lane < 16) ? 0 : 8;   // C-frag row base per ISA layout

    // ---- Phase 1: hidden = relu(x @ W_h + b_h); wave w owns cols 16w..16w+15
    // Depth-3 pipeline, 4-slot ring; sched_barrier(0) fences freeze the
    // "load i+3 before wmma i" order so waits stay partial.
    {
        v8f acc = {0.f, 0.f, 0.f, 0.f, 0.f, 0.f, 0.f, 0.f};
        const unsigned short* abase = xbf + (size_t)m * kD + aoff;
        const unsigned short* bbase =
            WhP + (size_t)w * (kD / 32) * 512 + (size_t)lane * 16;
        BFrag a[4], b[4];
        #pragma unroll
        for (int p = 0; p < kPipe; ++p) {
            const unsigned short* ap = abase + p * 32;
            a[p].q[0] = *(const uint4*)(ap);          // K {0..7} / {16..23}
            a[p].q[1] = *(const uint4*)(ap + 8);      // K {8..15} / {24..31}
            const uint4* bp = (const uint4*)(bbase + (size_t)p * 512);
            b[p].q[0] = bp[0];
            b[p].q[1] = bp[1];
        }
        __builtin_amdgcn_sched_barrier(0);
        #pragma unroll
        for (int kc = 0; kc < kD / 32; ++kc) {
            if (kc + kPipe < kD / 32) {
                const int nxt = (kc + kPipe) & 3;
                const unsigned short* ap = abase + (kc + kPipe) * 32;
                a[nxt].q[0] = *(const uint4*)(ap);
                a[nxt].q[1] = *(const uint4*)(ap + 8);
                const uint4* bp =
                    (const uint4*)(bbase + (size_t)(kc + kPipe) * 512);
                b[nxt].q[0] = bp[0];
                b[nxt].q[1] = bp[1];
            }
            __builtin_amdgcn_sched_barrier(0);   // loads(kc+3) stay before wmma(kc)
            const int cur = kc & 3;
            acc = __builtin_amdgcn_wmma_f32_16x16x32_bf16(
                false, a[cur].v, false, b[cur].v, (short)0, acc, false, false);
            __builtin_amdgcn_sched_barrier(0);
        }
        int colr = w * 16 + m;
        float bh = b_h[colr];
        #pragma unroll
        for (int r = 0; r < 8; ++r) {
            float hv = fmaxf(acc[r] + bh, 0.0f);
            hid[(size_t)(rbase + r) * kR + colr] = f32_to_bf16_rne(hv);
        }
    }
    __syncthreads();

    // ---- Phase 2: scores = hidden @ W_s + b_s; wave w owns 512 columns ------
    // Flattened (tt,kc) space, depth-3 pipelined B-fragment ring with fences.
    {
        BFrag A2[4];
        #pragma unroll
        for (int kc = 0; kc < 4; ++kc) {
            const unsigned short* ap = hid + (size_t)m * kR + kc * 32 + aoff;
            A2[kc].q[0] = *(const uint4*)(ap);
            A2[kc].q[1] = *(const uint4*)(ap + 8);
        }
        const unsigned short* bbase =
            WsP + (size_t)w * 32 * 4 * 512 + (size_t)lane * 16;
        BFrag bb[4];
        #pragma unroll
        for (int p = 0; p < kPipe; ++p) {
            const uint4* bp = (const uint4*)(bbase + (size_t)p * 512);
            bb[p].q[0] = bp[0];
            bb[p].q[1] = bp[1];
        }
        __builtin_amdgcn_sched_barrier(0);
        const v8f zacc = {0.f, 0.f, 0.f, 0.f, 0.f, 0.f, 0.f, 0.f};
        v8f acc = zacc;
        #pragma unroll
        for (int it = 0; it < 32 * 4; ++it) {
            const int tt = it >> 2;
            const int kc = it & 3;
            if (it + kPipe < 32 * 4) {
                const int nxt = (it + kPipe) & 3;
                const uint4* bp =
                    (const uint4*)(bbase + (size_t)(it + kPipe) * 512);
                bb[nxt].q[0] = bp[0];
                bb[nxt].q[1] = bp[1];
            }
            __builtin_amdgcn_sched_barrier(0);   // loads(it+3) stay before wmma(it)
            acc = __builtin_amdgcn_wmma_f32_16x16x32_bf16(
                false, A2[kc].v, false, bb[it & 3].v, (short)0, acc, false, false);
            __builtin_amdgcn_sched_barrier(0);
            if (kc == 3) {
                const int nt  = w * 32 + tt;
                const int col = nt * 16 + m;
                float bs = b_s[col];
                #pragma unroll
                for (int r = 0; r < 8; ++r)
                    scores[(size_t)(rbase + r) * kN + col] = acc[r] + bs;
                acc = zacc;
            }
        }
    }
    __syncthreads();

    // ---- Phase 3: per-token softmax stats (wave handles 2 tokens) -----------
    for (int rep = 0; rep < 2; ++rep) {
        int t = w * 2 + rep;
        const float* row = scores + (size_t)t * kN;
        float mx = -3.0e38f;
        for (int i = lane; i < kN; i += 32) mx = fmaxf(mx, row[i]);
        for (int off = 16; off > 0; off >>= 1) mx = fmaxf(mx, __shfl_xor(mx, off));
        float se = 0.0f;
        for (int i = lane; i < kN; i += 32) se += __expf(row[i] - mx);
        for (int off = 16; off > 0; off >>= 1) se += __shfl_xor(se, off);
        if (lane == 0) { tmax[t] = mx; tsum[t] = se; }
    }
    __syncthreads();

    // ---- Phase 3b: block-level avg-prob accumulation (1 atomic/slot/block) --
    for (int n = tid; n < kN; n += 256) {
        float a = 0.0f;
        for (int t = 0; t < kM; ++t)
            a += __expf(scores[(size_t)t * kN + n] - tmax[t]) / tsum[t];
        atomicAdd(avg_accum + n, a);
    }
    __syncthreads();

    // ---- Phase 4: top-32 with in-LDS knockout + budget-masked weights -------
    for (int rep = 0; rep < 2; ++rep) {
        int t = w * 2 + rep;
        float* row = scores + (size_t)t * kN;
        for (int j = 0; j < kMaxK; ++j) {
            float bv = -3.0e38f;
            int   bi = 0x7fffffff;
            for (int i = lane; i < kN; i += 32) {
                float v = row[i];
                if (v > bv || (v == bv && i < bi)) { bv = v; bi = i; }
            }
            for (int off = 16; off > 0; off >>= 1) {
                float ov = __shfl_xor(bv, off);
                int   oi = __shfl_xor(bi, off);
                if (ov > bv || (ov == bv && oi < bi)) { bv = ov; bi = oi; }
            }
            if (lane == 0) { topv[t * kMaxK + j] = bv; topil[t * kMaxK + j] = bi; }
            if ((bi & 31) == lane) row[bi] = -3.0e38f;   // knockout (in-order DS)
        }
        // lanes j = 0..31 compute the budget-masked softmax weights
        int   j   = lane;
        float v0  = topv[t * kMaxK + 0];
        float vj  = topv[t * kMaxK + j];
        int   ij  = topil[t * kMaxK + j];
        int   bud = budl[t];
        bool  act = j < bud;
        float e   = act ? __expf(vj - v0) : 0.0f;
        float s   = e;
        for (int off = 16; off > 0; off >>= 1) s += __shfl_xor(s, off);
        float wgt = act ? (e / s) : 0.0f;
        size_t go = (size_t)(t0 + t) * kMaxK + j;
        out_topi[go] = ij;
        out_w[go]    = wgt;
        if (act) atomicAdd(counts + ij, 1.0f);
    }
}

// ---------------------------------------------------------------------------
// Finalize: aux_loss = N * sum(avg_probs * avg_selection)
// ---------------------------------------------------------------------------
__global__ void finalize_aux(const float* __restrict__ avg,
                             const float* __restrict__ cnt,
                             float* __restrict__ out_aux) {
    __shared__ float r[256];
    const float inv = 1.0f / (float)kTokens;
    float s = 0.0f;
    for (int n = threadIdx.x; n < kN; n += 256)
        s += (avg[n] * inv) * (cnt[n] * inv);
    r[threadIdx.x] = s;
    __syncthreads();
    for (int off = 128; off > 0; off >>= 1) {
        if (threadIdx.x < off) r[threadIdx.x] += r[threadIdx.x + off];
        __syncthreads();
    }
    if (threadIdx.x == 0) out_aux[0] = (float)kN * r[0];
}

// ---------------------------------------------------------------------------
extern "C" void kernel_launch(void* const* d_in, const int* in_sizes, int n_in,
                              void* d_out, int out_size, void* d_ws, size_t ws_size,
                              hipStream_t stream) {
    const float* x   = (const float*)d_in[0];
    const float* W_c = (const float*)d_in[1];
    const float* b_c = (const float*)d_in[2];
    const float* W_h = (const float*)d_in[3];
    const float* b_h = (const float*)d_in[4];
    const float* W_s = (const float*)d_in[5];
    const float* b_s = (const float*)d_in[6];

    // workspace carve: packed bf16 weights + global accumulators (~1.3 MB)
    char* ws = (char*)d_ws;
    unsigned short* WhP = (unsigned short*)ws;                       // 131072 halfs
    unsigned short* WsP = (unsigned short*)(ws + 262144);            // 524288 halfs
    float* avg_accum    = (float*)(ws + 262144 + 1048576);           // 4096 f32
    float* counts       = avg_accum + 4096;                          // 4096 f32

    // output carve: top_i | weights | budget | aux_loss
    int*   out_topi = (int*)d_out;                       // 32768*32
    float* out_w    = (float*)d_out + 1048576;           // 32768*32
    int*   out_bud  = (int*)d_out + 2097152;             // 32768
    float* out_aux  = (float*)d_out + 2129920;           // 1

    hipLaunchKernelGGL(zero_accum, dim3(32), dim3(256), 0, stream,
                       avg_accum, 8192);
    hipLaunchKernelGGL(pack_b_frags, dim3(512), dim3(256), 0, stream,
                       W_h, WhP, kR, kD / 32, kD * kR);
    hipLaunchKernelGGL(pack_b_frags, dim3(2048), dim3(256), 0, stream,
                       W_s, WsP, kN, kR / 32, kR * kN);

    (void)hipFuncSetAttribute((const void*)router_main,
                              hipFuncAttributeMaxDynamicSharedMemorySize,
                              kSmemBytes);
    hipLaunchKernelGGL(router_main, dim3(kTokens / kM), dim3(256), kSmemBytes,
                       stream, x, W_c, b_c, b_h, b_s, WhP, WsP,
                       avg_accum, counts, out_topi, out_w, out_bud);

    hipLaunchKernelGGL(finalize_aux, dim3(1), dim3(256), 0, stream,
                       avg_accum, counts, out_aux);
}
